// Forward_warp_guided_no_pcd_48223892800162
// MI455X (gfx1250) — matrix operand order repaired and checked by reading, not compiled, not run on hardware
//
#include <hip/hip_runtime.h>

typedef _Float16 v16h __attribute__((ext_vector_type(16)));
typedef _Float16 v8h  __attribute__((ext_vector_type(8)));
typedef float    v8f  __attribute__((ext_vector_type(8)));

#define ALPHA_M (-20.0f)

static inline int cdivl(long a, long b) { return (int)((a + b - 1) / b); }

// ---------------------------------------------------------------------------
// Elementwise / data-movement kernels
// ---------------------------------------------------------------------------

__global__ void k_zero(float* __restrict__ p, long n) {
  long i = (long)blockIdx.x * blockDim.x + threadIdx.x;
  if (i < n) p[i] = 0.0f;
}

__global__ void k_add_inplace(float* __restrict__ o, const float* __restrict__ a, long n) {
  long i = (long)blockIdx.x * blockDim.x + threadIdx.x;
  if (i < n) o[i] += a[i];
}

// anchor = 0.5 * (where(w1>0,w1,w2) + where(w2>0,w2,w1))
__global__ void k_merge(float* __restrict__ o, const float* __restrict__ w1,
                        const float* __restrict__ w2, long n) {
  long i = (long)blockIdx.x * blockDim.x + threadIdx.x;
  if (i >= n) return;
  float a = w1[i], b = w2[i];
  float av = (a > 0.0f) ? a : b;
  float bv = (b > 0.0f) ? b : a;
  o[i] = 0.5f * (av + bv);
}

// hs = mk * sa + (1-mk) * sb   (mk is 1-channel broadcast)
__global__ void k_blend(float* __restrict__ o, const float* __restrict__ mk,
                        const float* __restrict__ sa, const float* __restrict__ sb,
                        int B, int C, int H, int W) {
  long i = (long)blockIdx.x * blockDim.x + threadIdx.x;
  long tot = (long)B * C * H * W;
  if (i >= tot) return;
  long hw = (long)H * W;
  int x = (int)(i % W); long p = i / W;
  int y = (int)(p % H); p /= H;
  p /= C;                       // skip c
  int b = (int)p;
  float m = mk[(long)b * hw + (long)y * W + x];
  o[i] = m * sa[i] + (1.0f - m) * sb[i];
}

// generic bilinear resize, same half-pixel/clamp convention as reference
__global__ void k_resize(float* __restrict__ o, const float* __restrict__ x,
                         int B, int C, int Hi, int Wi, int Ho, int Wo) {
  long i = (long)blockIdx.x * blockDim.x + threadIdx.x;
  long tot = (long)B * C * Ho * Wo;
  if (i >= tot) return;
  int ox = (int)(i % Wo); long p = i / Wo;
  int oy = (int)(p % Ho); p /= Ho;
  int c = (int)(p % C);   int b = (int)(p / C);
  float cy = (oy + 0.5f) * ((float)Hi / (float)Ho) - 0.5f;
  cy = cy < 0.0f ? 0.0f : (cy > (float)(Hi - 1) ? (float)(Hi - 1) : cy);
  int y0 = (int)floorf(cy); int y1 = (y0 + 1 < Hi) ? y0 + 1 : Hi - 1;
  float wy = cy - (float)y0;
  float cx = (ox + 0.5f) * ((float)Wi / (float)Wo) - 0.5f;
  cx = cx < 0.0f ? 0.0f : (cx > (float)(Wi - 1) ? (float)(Wi - 1) : cx);
  int x0 = (int)floorf(cx); int x1 = (x0 + 1 < Wi) ? x0 + 1 : Wi - 1;
  float wx = cx - (float)x0;
  const float* xc = x + ((long)b * C + c) * (long)Hi * Wi;
  float v00 = xc[(long)y0 * Wi + x0], v01 = xc[(long)y0 * Wi + x1];
  float v10 = xc[(long)y1 * Wi + x0], v11 = xc[(long)y1 * Wi + x1];
  o[i] = (1.0f - wy) * ((1.0f - wx) * v00 + wx * v01) + wy * ((1.0f - wx) * v10 + wx * v11);
}

// m = scale * mean_c | a - backwarp(bimg, flow) |
__global__ void k_backwarp_metric(float* __restrict__ m, const float* __restrict__ a,
                                  const float* __restrict__ bimg, const float* __restrict__ flow,
                                  const float* __restrict__ scaleptr,
                                  int B, int C, int H, int W) {
  long i = (long)blockIdx.x * blockDim.x + threadIdx.x;
  long tot = (long)B * H * W;
  if (i >= tot) return;
  int x = (int)(i % W); long p = i / W;
  int y = (int)(p % H); int b = (int)(p / H);
  long hw = (long)H * W;
  const float* fb = flow + (long)b * 2 * hw;
  float gx = (float)x + fb[(long)y * W + x];
  float gy = (float)y + fb[hw + (long)y * W + x];
  int x0 = (int)floorf(gx), y0 = (int)floorf(gy);
  float sum = 0.0f;
  for (int c = 0; c < C; ++c) {
    const float* xc = bimg + ((long)b * C + c) * hw;
    float g = 0.0f;
    for (int dy = 0; dy < 2; ++dy)
      for (int dx = 0; dx < 2; ++dx) {
        int xi = x0 + dx, yi = y0 + dy;
        float wgt = (1.0f - fabsf(gx - (float)xi)) * (1.0f - fabsf(gy - (float)yi));
        if (xi >= 0 && xi < W && yi >= 0 && yi < H)
          g += xc[(long)yi * W + xi] * wgt;
      }
    sum += fabsf(a[((long)b * C + c) * hw + (long)y * W + x] - g);
  }
  m[i] = scaleptr[0] * (sum / (float)C);
}

// softsplat scatter: acc[(b,c,yi,xi)] += inp*exp(ALPHA*metric)*w ; den channel C
__global__ void k_splat_scatter(const float* __restrict__ inp, const float* __restrict__ flow,
                                const float* __restrict__ metric, const float* __restrict__ tptr,
                                int oneMinus, float* __restrict__ acc,
                                int C, int B, int H, int W) {
  long i = (long)blockIdx.x * blockDim.x + threadIdx.x;
  long tot = (long)B * H * W;
  if (i >= tot) return;
  int x = (int)(i % W); long p = i / W;
  int y = (int)(p % H); int b = (int)(p / H);
  long hw = (long)H * W;
  float f = tptr[0];
  if (oneMinus) f = 1.0f - f;
  const float* fb = flow + (long)b * 2 * hw;
  float gx = (float)x + f * fb[(long)y * W + x];
  float gy = (float)y + f * fb[hw + (long)y * W + x];
  float e = expf(ALPHA_M * metric[(long)b * hw + (long)y * W + x]);
  int x0 = (int)floorf(gx), y0 = (int)floorf(gy);
  float* accb = acc + (long)b * (C + 1) * hw;
  const float* inb = inp + (long)b * C * hw;
  long spix = (long)y * W + x;
  for (int dy = 0; dy < 2; ++dy)
    for (int dx = 0; dx < 2; ++dx) {
      int xi = x0 + dx, yi = y0 + dy;
      if (xi < 0 || xi >= W || yi < 0 || yi >= H) continue;
      float wgt = (1.0f - fabsf(gx - (float)xi)) * (1.0f - fabsf(gy - (float)yi)) * e;
      long di = (long)yi * W + xi;
      for (int c = 0; c < C; ++c)
        atomicAdd(accb + (long)c * hw + di, inb[(long)c * hw + spix] * wgt);
      atomicAdd(accb + (long)C * hw + di, wgt);
    }
}

__global__ void k_splat_norm(const float* __restrict__ acc, float* __restrict__ o,
                             int C, int B, int H, int W) {
  long i = (long)blockIdx.x * blockDim.x + threadIdx.x;
  long tot = (long)B * C * H * W;
  if (i >= tot) return;
  long hw = (long)H * W;
  int x = (int)(i % W); long p = i / W;
  int y = (int)(p % H); p /= H;
  int c = (int)(p % C); int b = (int)(p / C);
  long base = (long)b * (C + 1) * hw;
  float den = acc[base + (long)C * hw + (long)y * W + x];
  if (den == 0.0f) den = 1.0f;
  o[i] = acc[base + (long)c * hw + (long)y * W + x] / den;
}

// NCHW f32 (up to 3 concatenated sources) -> NHWC f16 with channel pad to Cpad
__global__ void k_pack_nhwc(_Float16* __restrict__ dst,
                            const float* __restrict__ x0, int c0,
                            const float* __restrict__ x1, int c1,
                            const float* __restrict__ x2, int c2,
                            int Cpad, int B, int H, int W) {
  long i = (long)blockIdx.x * blockDim.x + threadIdx.x;
  long tot = (long)B * H * W * Cpad;
  if (i >= tot) return;
  int c = (int)(i % Cpad); long p = i / Cpad;
  int x = (int)(p % W); p /= W;
  int y = (int)(p % H); int b = (int)(p / H);
  long hw = (long)H * W; long pi = (long)y * W + x;
  float v = 0.0f;
  if (c < c0)                v = x0[((long)b * c0 + c) * hw + pi];
  else if (c < c0 + c1)      v = x1[((long)b * c1 + (c - c0)) * hw + pi];
  else if (c < c0 + c1 + c2) v = x2[((long)b * c2 + (c - c0 - c1)) * hw + pi];
  dst[i] = (_Float16)v;
}

// OIHW f32 -> f16 wp[n][tap*Cpad + c], zero padded to Npad x 9*Cpad
__global__ void k_pack_w(_Float16* __restrict__ wp, const float* __restrict__ w,
                         int Cout, int Cin, int Cpad, int Npad) {
  long K = 9L * Cpad;
  long i = (long)blockIdx.x * blockDim.x + threadIdx.x;
  long tot = (long)Npad * K;
  if (i >= tot) return;
  int n = (int)(i / K); long r = i - (long)n * K;
  int tap = (int)(r / Cpad); int c = (int)(r - (long)tap * Cpad);
  float v = 0.0f;
  if (n < Cout && c < Cin) v = w[((long)n * Cin + c) * 9 + tap];
  wp[i] = (_Float16)v;
}

// ---------------------------------------------------------------------------
// WMMA implicit-GEMM 3x3 conv, pad=1, fused bias + activation.
//   Template: CPT = Cpad/32 (K chunks per filter tap), NT = Npad/16 N-tiles,
//   STRIDE in {1,2}. 8 waves/block, each wave owns 2 M-tiles x 1 N-tile ->
//   2 v_wmma_f32_16x16x32_f16 per K-chunk on independent accumulators.
//   LDS ping-pong: stage chunk k+1 while computing chunk k, 1 barrier/chunk.
//   xh: NHWC f16 [B,Hin,Win,Cpad]  wp: f16 [Npad][9*Cpad]  out: NCHW f32
// ---------------------------------------------------------------------------
template<int CPT, int NT, int STRIDE>
__global__ __launch_bounds__(256, 1)
void k_conv_wmma(const _Float16* __restrict__ xh, const _Float16* __restrict__ wp,
                 const float* __restrict__ bias, float* __restrict__ out,
                 int B, int Hin, int Win, int Hout, int Wout, int Cout, int act) {
  constexpr int Cpad = CPT * 32;
  constexpr int K = 9 * Cpad;
  constexpr int nChunks = 9 * CPT;
  constexpr int Mblock = (8 / NT) * 32;        // 2 M-tiles per wave
  constexpr int SIters = (Mblock * 4) / 256;   // v8h staging slots per thread
  __shared__ _Float16 As[2][Mblock * 32];

  const int tid = threadIdx.x;
  const int lane = tid & 31;
  const int wave = tid >> 5;
  const int ni  = wave % NT;
  const int mi0 = (wave / NT) * 2;
  const long Mtotal = (long)B * Hout * Wout;
  const long mb = (long)blockIdx.x * Mblock;

  // chunk-invariant staging coordinates (hoisted out of the K loop)
  int st_oy[SIters], st_ox[SIters], st_b[SIters], st_row[SIters], st_seg[SIters];
  bool st_valid[SIters];
  #pragma unroll
  for (int si = 0; si < SIters; ++si) {
    int s = tid + si * 256;
    st_row[si] = s >> 2; st_seg[si] = s & 3;
    long m = mb + (s >> 2);
    bool v = m < Mtotal;
    long mm = v ? m : 0;
    st_ox[si] = (int)(mm % Wout);
    long p2 = mm / Wout;
    st_oy[si] = (int)(p2 % Hout);
    st_b[si]  = (int)(p2 / Hout);
    st_valid[si] = v;
  }

  auto stage = [&](int kc, int buf) {
    const int tap  = kc / CPT;               // CPT is constexpr -> cheap
    const int cin0 = (kc - tap * CPT) << 5;
    const int ky = tap / 3 - 1, kx = tap % 3 - 1;
    #pragma unroll
    for (int si = 0; si < SIters; ++si) {
      v8h v = {};
      const int iy = st_oy[si] * STRIDE + ky;
      const int ix = st_ox[si] * STRIDE + kx;
      if (st_valid[si] && iy >= 0 && iy < Hin && ix >= 0 && ix < Win)
        v = *(const v8h*)(xh + (((long)st_b[si] * Hin + iy) * Win + ix) * Cpad
                          + cin0 + st_seg[si] * 8);
      *(v8h*)(&As[buf][st_row[si] * 32 + st_seg[si] * 8]) = v;
    }
  };

  v8f acc0 = {}, acc1 = {};
  stage(0, 0);
  __syncthreads();

  // A lane layout (16-bit A 16x32): lane<16 -> K{0..7,16..23}, lane>=16 -> K{8..15,24..31}
  const int arow0 = (mi0 * 16 + (lane & 15)) * 32;
  const int arow1 = arow0 + 16 * 32;
  const int kb = (lane >> 4) << 3;
  // B lane layout: column n = lane%16, lane<16 -> K{0..7,16..23}, lane>=16 -> K{8..15,24..31}
  const _Float16* bbase = wp + (long)(ni * 16 + (lane & 15)) * K + ((lane >> 4) << 3);

  for (int kc = 0; kc < nChunks; ++kc) {
    const int cur = kc & 1;
    if (kc + 1 < nChunks) {
      __builtin_prefetch(bbase + (kc + 1) * 32, 0, 1);   // global_prefetch_b8
      stage(kc + 1, cur ^ 1);
    }
    v8h a00 = *(const v8h*)(&As[cur][arow0 + kb]);
    v8h a01 = *(const v8h*)(&As[cur][arow0 + kb + 16]);
    v8h a10 = *(const v8h*)(&As[cur][arow1 + kb]);
    v8h a11 = *(const v8h*)(&As[cur][arow1 + kb + 16]);
    v16h a0 = __builtin_shufflevector(a00, a01, 0, 1, 2, 3, 4, 5, 6, 7,
                                      8, 9, 10, 11, 12, 13, 14, 15);
    v16h a1 = __builtin_shufflevector(a10, a11, 0, 1, 2, 3, 4, 5, 6, 7,
                                      8, 9, 10, 11, 12, 13, 14, 15);
    v16h bm = __builtin_shufflevector(*(const v8h*)(bbase + kc * 32),
                                      *(const v8h*)(bbase + kc * 32 + 16),
                                      0, 1, 2, 3, 4, 5, 6, 7, 8, 9, 10, 11, 12, 13, 14, 15);
    acc0 = __builtin_amdgcn_wmma_f32_16x16x32_f16(false, a0, false, bm,
                                                  (short)0, acc0, false, false);
    acc1 = __builtin_amdgcn_wmma_f32_16x16x32_f16(false, a1, false, bm,
                                                  (short)0, acc1, false, false);
    __syncthreads();
  }

  // C/D layout: lane col n = lane%16; VGPR r -> row r + 8*(lane/16)
  const int cout = ni * 16 + (lane & 15);
  if (cout < Cout) {
    const float bv = bias[cout];
    const long hw = (long)Hout * Wout;
    #pragma unroll
    for (int half = 0; half < 2; ++half) {
      v8f acc;
      if (half == 0) acc = acc0; else acc = acc1;
      const long mrow = mb + (mi0 + half) * 16 + ((lane >> 4) << 3);
      for (int r = 0; r < 8; ++r) {
        long m = mrow + r;
        if (m < Mtotal) {
          float v = acc[r] + bv;
          if (act == 1)      v = (v >= 0.0f) ? v : 0.1f * v;     // lrelu
          else if (act == 2) v = 1.0f / (1.0f + expf(-v));       // sigmoid
          long b = m / hw; long pix = m - b * hw;
          out[(b * Cout + cout) * hw + pix] = v;
        }
      }
    }
  }
}

// ---------------------------------------------------------------------------
// Host-side helpers
// ---------------------------------------------------------------------------

static void run_conv(hipStream_t s,
                     const float* x0, int c0, const float* x1, int c1,
                     const float* x2, int c2,
                     int Bn, int Hin, int Win, int stride,
                     const float* wO, const float* bias, int Cout, int act,
                     _Float16* xh, _Float16* wpk, float* out) {
  const int Cin = c0 + c1 + c2;
  const int Cpad = ((Cin + 31) / 32) * 32;
  const int Npad = ((Cout + 15) / 16) * 16;
  const long nE = (long)Bn * Hin * Win * Cpad;
  k_pack_nhwc<<<cdivl(nE, 256), 256, 0, s>>>(xh, x0, c0, x1, c1, x2, c2, Cpad, Bn, Hin, Win);
  const long wE = (long)Npad * 9 * Cpad;
  k_pack_w<<<cdivl(wE, 256), 256, 0, s>>>(wpk, wO, Cout, Cin, Cpad, Npad);
  const int Hout = (Hin - 1) / stride + 1;
  const int Wout = (Win - 1) / stride + 1;
  const long Mtotal = (long)Bn * Hout * Wout;
  const int CPT = Cpad >> 5;
  const int NT = Npad >> 4;
  const int Mblock = (8 / NT) * 32;
  const int blocks = cdivl(Mtotal, Mblock);
#define LC(cpt, nt, st) \
  k_conv_wmma<cpt, nt, st><<<blocks, 256, 0, s>>>(xh, wpk, bias, out, Bn, Hin, Win, \
                                                  Hout, Wout, Cout, act)
  if (stride == 2) {
    LC(2, 4, 2);                       // 64ch stride-2 blocks
  } else if (NT == 1) {
    LC(4, 1, 1);                       // 128ch -> 1ch mask convs
  } else {
    switch (CPT) {
      case 1: LC(1, 4, 1); break;      // fea1: 3 -> 64
      case 2: LC(2, 4, 1); break;      // 64 -> 64
      case 4: LC(4, 4, 1); break;      // red1: 128 -> 64
      case 5: LC(5, 4, 1); break;      // red3 / sc: 131 -> 64
      case 6: LC(6, 4, 1); break;      // red2: 192 -> 64
      default: break;
    }
  }
#undef LC
}

static void run_splat(hipStream_t s, const float* inp, int C, int Bn, int H, int W,
                      const float* flow, const float* metric, const float* tptr,
                      int oneMinus, float* acc, float* out) {
  const long accN = (long)Bn * (C + 1) * H * W;
  k_zero<<<cdivl(accN, 256), 256, 0, s>>>(acc, accN);
  const long pix = (long)Bn * H * W;
  k_splat_scatter<<<cdivl(pix, 256), 256, 0, s>>>(inp, flow, metric, tptr, oneMinus,
                                                  acc, C, Bn, H, W);
  const long oN = (long)Bn * C * H * W;
  k_splat_norm<<<cdivl(oN, 256), 256, 0, s>>>(acc, out, C, Bn, H, W);
}

static void run_resize(hipStream_t s, float* o, const float* x,
                       int B, int C, int Hi, int Wi, int Ho, int Wo) {
  const long n = (long)B * C * Ho * Wo;
  k_resize<<<cdivl(n, 256), 256, 0, s>>>(o, x, B, C, Hi, Wi, Ho, Wo);
}

// ---------------------------------------------------------------------------

extern "C" void kernel_launch(void* const* d_in, const int* in_sizes, int n_in,
                              void* d_out, int out_size, void* d_ws, size_t ws_size,
                              hipStream_t stream) {
  (void)in_sizes; (void)n_in; (void)out_size; (void)ws_size;
  const int Bn = 2, C = 64, H = 256, W = 256;
  const int H2 = 128, W2 = 128, H3 = 64, W3 = 64;
  const long hw = (long)H * W, hw2 = (long)H2 * W2, hw3 = (long)H3 * W3;

  const float* img1 = (const float*)d_in[0];
  const float* img2 = (const float*)d_in[1];
  const float* feat1 = (const float*)d_in[2];
  const float* feat2 = (const float*)d_in[3];
  const float* f11 = (const float*)d_in[4];
  const float* f12 = (const float*)d_in[5];
  const float* f13 = (const float*)d_in[6];
  const float* f21 = (const float*)d_in[7];
  const float* f22 = (const float*)d_in[8];
  const float* f23 = (const float*)d_in[9];
  const float* tptr = (const float*)d_in[10];
  // params flattened in dict order
  const float* p_scale  = (const float*)d_in[11];
  const float* fea1_w   = (const float*)d_in[12]; const float* fea1_b   = (const float*)d_in[13];
  const float* feaL2_w1 = (const float*)d_in[14]; const float* feaL2_b1 = (const float*)d_in[15];
  const float* feaL2_w2 = (const float*)d_in[16]; const float* feaL2_b2 = (const float*)d_in[17];
  const float* feaL3_w1 = (const float*)d_in[18]; const float* feaL3_b1 = (const float*)d_in[19];
  const float* feaL3_w2 = (const float*)d_in[20]; const float* feaL3_b2 = (const float*)d_in[21];
  const float* hsL2_w1  = (const float*)d_in[22]; const float* hsL2_b1  = (const float*)d_in[23];
  const float* hsL2_w2  = (const float*)d_in[24]; const float* hsL2_b2  = (const float*)d_in[25];
  const float* hsL3_w1  = (const float*)d_in[26]; const float* hsL3_b1  = (const float*)d_in[27];
  const float* hsL3_w2  = (const float*)d_in[28]; const float* hsL3_b2  = (const float*)d_in[29];
  const float* red1_w   = (const float*)d_in[30]; const float* red1_b   = (const float*)d_in[31];
  const float* red2_w   = (const float*)d_in[32]; const float* red2_b   = (const float*)d_in[33];
  const float* red3_w   = (const float*)d_in[34]; const float* red3_b   = (const float*)d_in[35];
  const float* m3_w     = (const float*)d_in[36]; const float* m3_b     = (const float*)d_in[37];
  const float* m2_w     = (const float*)d_in[38]; const float* m2_b     = (const float*)d_in[39];
  const float* m1_w     = (const float*)d_in[40]; const float* m1_b     = (const float*)d_in[41];
  const float* sc_w     = (const float*)d_in[42]; const float* sc_b     = (const float*)d_in[43];
  float* out = (float*)d_out;

  // ---- bump arena in d_ws ----
  char* base = (char*)d_ws; size_t off = 0;
  auto A = [&](long elems) -> float* {
    float* p = (float*)(base + off);
    off = (off + (size_t)elems * 4 + 255) & ~(size_t)255;
    return p;
  };
  auto Ah = [&](long elems) -> _Float16* {
    _Float16* p = (_Float16*)(base + off);
    off = (off + (size_t)elems * 2 + 255) & ~(size_t)255;
    return p;
  };

  float* m12   = A(Bn * hw);  float* m21   = A(Bn * hw);
  float* m12_2 = A(Bn * hw2); float* m21_2 = A(Bn * hw2);
  float* m12_3 = A(Bn * hw3); float* m21_3 = A(Bn * hw3);
  float* accA = A((long)Bn * 65 * hw);   // splat accumulators, reused at all levels
  float* accB = A((long)Bn * 65 * hw);
  float* wc1 = A((long)Bn * 3 * hw); float* wc2 = A((long)Bn * 3 * hw);
  float* anchor_img = A((long)Bn * 3 * hw);
  float* L1   = A((long)Bn * C * hw);    // reused for img1 then img2
  float* tmpF = A((long)Bn * C * hw);    // conv intermediates
  float* tmpH = A((long)Bn * C * hw2);
  float* L2f1 = A((long)Bn * C * hw2); float* L2f2 = A((long)Bn * C * hw2);
  float* L3f1 = A((long)Bn * C * hw3); float* L3f2 = A((long)Bn * C * hw3);
  float* h2a = A((long)Bn * C * hw2); float* h2b = A((long)Bn * C * hw2);
  float* h3a = A((long)Bn * C * hw3); float* h3b = A((long)Bn * C * hw3);
  float* wa1 = A((long)Bn * C * hw2); float* wa2 = A((long)Bn * C * hw2);
  float* wb1 = A((long)Bn * C * hw3); float* wb2 = A((long)Bn * C * hw3);
  float* anchor_l2 = A((long)Bn * C * hw2);
  float* anchor_l3 = A((long)Bn * C * hw3);
  float* s3a = A((long)Bn * C * hw3); float* s3b = A((long)Bn * C * hw3);
  float* s2a = A((long)Bn * C * hw2); float* s2b = A((long)Bn * C * hw2);
  float* s1a = A((long)Bn * C * hw);  float* s1b = A((long)Bn * C * hw);
  float* mk3 = A(Bn * hw3); float* mk2 = A(Bn * hw2); float* mk1 = A(Bn * hw);
  float* bl3 = A((long)Bn * C * hw3);   // blended hs3 before red1
  float* bl2 = A((long)Bn * C * hw2);
  float* bl1 = A((long)Bn * C * hw);
  float* hs3q = A((long)Bn * C * hw3); float* hs3 = A((long)Bn * C * hw2);
  float* hs2h = A((long)Bn * C * hw2); float* hs2 = A((long)Bn * C * hw);
  float* hs1  = A((long)Bn * C * hw);
  float* a2  = A((long)Bn * C * hw);
  float* a3h = A((long)Bn * C * hw2);
  float* a3  = A((long)Bn * C * hw);
  _Float16* xh  = Ah((long)Bn * hw * 160);  // NHWC f16 staging (max Cpad at full res)
  _Float16* wpk = Ah(131072);               // packed f16 weights (max 64*9*192)

  // ---- 1. occlusion metrics + pyramids ----
  {
    long n = (long)Bn * hw;
    k_backwarp_metric<<<cdivl(n, 256), 256, 0, stream>>>(m12, img1, img2, f11, p_scale, Bn, 3, H, W);
    k_backwarp_metric<<<cdivl(n, 256), 256, 0, stream>>>(m21, img2, img1, f21, p_scale, Bn, 3, H, W);
  }
  run_resize(stream, m12_2, m12, Bn, 1, H, W, H2, W2);
  run_resize(stream, m12_3, m12, Bn, 1, H, W, H3, W3);
  run_resize(stream, m21_2, m21, Bn, 1, H, W, H2, W2);
  run_resize(stream, m21_3, m21, Bn, 1, H, W, H3, W3);

  // ---- 2. warped color anchor ----
  run_splat(stream, img1, 3, Bn, H, W, f11, m12, tptr, 0, accA, wc1);
  run_splat(stream, img2, 3, Bn, H, W, f21, m21, tptr, 1, accB, wc2);
  k_merge<<<cdivl((long)Bn * 3 * hw, 256), 256, 0, stream>>>(anchor_img, wc1, wc2, (long)Bn * 3 * hw);

  // ---- 3. shallow feature pyramids (WMMA convs) ----
  // image 1
  run_conv(stream, img1, 3, nullptr, 0, nullptr, 0, Bn, H, W, 1, fea1_w, fea1_b, C, 1, xh, wpk, L1);
  run_conv(stream, L1, C, nullptr, 0, nullptr, 0, Bn, H, W, 1, feaL2_w1, feaL2_b1, C, 1, xh, wpk, tmpF);
  run_conv(stream, tmpF, C, nullptr, 0, nullptr, 0, Bn, H, W, 2, feaL2_w2, feaL2_b2, C, 1, xh, wpk, L2f1);
  run_conv(stream, L2f1, C, nullptr, 0, nullptr, 0, Bn, H2, W2, 1, feaL3_w1, feaL3_b1, C, 1, xh, wpk, tmpH);
  run_conv(stream, tmpH, C, nullptr, 0, nullptr, 0, Bn, H2, W2, 2, feaL3_w2, feaL3_b2, C, 1, xh, wpk, L3f1);
  // image 2
  run_conv(stream, img2, 3, nullptr, 0, nullptr, 0, Bn, H, W, 1, fea1_w, fea1_b, C, 1, xh, wpk, L1);
  run_conv(stream, L1, C, nullptr, 0, nullptr, 0, Bn, H, W, 1, feaL2_w1, feaL2_b1, C, 1, xh, wpk, tmpF);
  run_conv(stream, tmpF, C, nullptr, 0, nullptr, 0, Bn, H, W, 2, feaL2_w2, feaL2_b2, C, 1, xh, wpk, L2f2);
  run_conv(stream, L2f2, C, nullptr, 0, nullptr, 0, Bn, H2, W2, 1, feaL3_w1, feaL3_b1, C, 1, xh, wpk, tmpH);
  run_conv(stream, tmpH, C, nullptr, 0, nullptr, 0, Bn, H2, W2, 2, feaL3_w2, feaL3_b2, C, 1, xh, wpk, L3f2);

  // ---- 4. warped feature anchors ----
  run_splat(stream, L2f1, C, Bn, H2, W2, f12, m12_2, tptr, 0, accA, wa1);
  run_splat(stream, L2f2, C, Bn, H2, W2, f22, m21_2, tptr, 1, accB, wa2);
  k_merge<<<cdivl((long)Bn * C * hw2, 256), 256, 0, stream>>>(anchor_l2, wa1, wa2, (long)Bn * C * hw2);
  run_splat(stream, L3f1, C, Bn, H3, W3, f13, m12_3, tptr, 0, accA, wb1);
  run_splat(stream, L3f2, C, Bn, H3, W3, f23, m21_3, tptr, 1, accB, wb2);
  k_merge<<<cdivl((long)Bn * C * hw3, 256), 256, 0, stream>>>(anchor_l3, wb1, wb2, (long)Bn * C * hw3);

  // ---- 5. deep feature pyramid (hs) ----
  run_conv(stream, feat1, C, nullptr, 0, nullptr, 0, Bn, H, W, 1, hsL2_w1, hsL2_b1, C, 1, xh, wpk, tmpF);
  run_conv(stream, tmpF, C, nullptr, 0, nullptr, 0, Bn, H, W, 2, hsL2_w2, hsL2_b2, C, 1, xh, wpk, h2a);
  run_conv(stream, feat2, C, nullptr, 0, nullptr, 0, Bn, H, W, 1, hsL2_w1, hsL2_b1, C, 1, xh, wpk, tmpF);
  run_conv(stream, tmpF, C, nullptr, 0, nullptr, 0, Bn, H, W, 2, hsL2_w2, hsL2_b2, C, 1, xh, wpk, h2b);
  run_conv(stream, h2a, C, nullptr, 0, nullptr, 0, Bn, H2, W2, 1, hsL3_w1, hsL3_b1, C, 1, xh, wpk, tmpH);
  run_conv(stream, tmpH, C, nullptr, 0, nullptr, 0, Bn, H2, W2, 2, hsL3_w2, hsL3_b2, C, 1, xh, wpk, h3a);
  run_conv(stream, h2b, C, nullptr, 0, nullptr, 0, Bn, H2, W2, 1, hsL3_w1, hsL3_b1, C, 1, xh, wpk, tmpH);
  run_conv(stream, tmpH, C, nullptr, 0, nullptr, 0, Bn, H2, W2, 2, hsL3_w2, hsL3_b2, C, 1, xh, wpk, h3b);

  // ---- 6. level 3 fusion ----
  run_splat(stream, h3a, C, Bn, H3, W3, f13, m12_3, tptr, 0, accA, s3a);
  run_splat(stream, h3b, C, Bn, H3, W3, f23, m21_3, tptr, 1, accB, s3b);
  run_conv(stream, s3a, C, s3b, C, nullptr, 0, Bn, H3, W3, 1, m3_w, m3_b, 1, 2, xh, wpk, mk3);
  k_blend<<<cdivl((long)Bn * C * hw3, 256), 256, 0, stream>>>(bl3, mk3, s3a, s3b, Bn, C, H3, W3);
  run_conv(stream, anchor_l3, C, bl3, C, nullptr, 0, Bn, H3, W3, 1, red1_w, red1_b, C, 0, xh, wpk, hs3q);
  run_resize(stream, hs3, hs3q, Bn, C, H3, W3, H2, W2);

  // ---- 7. level 2 fusion ----
  run_splat(stream, h2a, C, Bn, H2, W2, f12, m12_2, tptr, 0, accA, s2a);
  run_splat(stream, h2b, C, Bn, H2, W2, f22, m21_2, tptr, 1, accB, s2b);
  run_conv(stream, s2a, C, s2b, C, nullptr, 0, Bn, H2, W2, 1, m2_w, m2_b, 1, 2, xh, wpk, mk2);
  k_blend<<<cdivl((long)Bn * C * hw2, 256), 256, 0, stream>>>(bl2, mk2, s2a, s2b, Bn, C, H2, W2);
  run_conv(stream, anchor_l2, C, hs3, C, bl2, C, Bn, H2, W2, 1, red2_w, red2_b, C, 0, xh, wpk, hs2h);
  run_resize(stream, hs2, hs2h, Bn, C, H2, W2, H, W);

  // ---- 8. level 1 fusion ----
  run_splat(stream, feat1, C, Bn, H, W, f11, m12, tptr, 0, accA, s1a);
  run_splat(stream, feat2, C, Bn, H, W, f21, m21, tptr, 1, accB, s1b);
  run_conv(stream, s1a, C, s1b, C, nullptr, 0, Bn, H, W, 1, m1_w, m1_b, 1, 2, xh, wpk, mk1);
  k_blend<<<cdivl((long)Bn * C * hw, 256), 256, 0, stream>>>(bl1, mk1, s1a, s1b, Bn, C, H, W);
  run_conv(stream, anchor_img, 3, hs2, C, bl1, C, Bn, H, W, 1, red3_w, red3_b, C, 0, xh, wpk, hs1);

  // ---- 9. anchor synthesis + output ----
  run_resize(stream, a2, anchor_l2, Bn, C, H2, W2, H, W);
  run_resize(stream, a3h, anchor_l3, Bn, C, H3, W3, H2, W2);
  run_resize(stream, a3, a3h, Bn, C, H2, W2, H, W);
  run_conv(stream, anchor_img, 3, a2, C, a3, C, Bn, H, W, 1, sc_w, sc_b, C, 1, xh, wpk, out);
  k_add_inplace<<<cdivl((long)Bn * C * hw, 256), 256, 0, stream>>>(out, hs1, (long)Bn * C * hw);
}
